// Linear_38225208934974
// MI455X (gfx1250) — matrix operand order repaired and checked by reading, not compiled, not run on hardware
//
#include <hip/hip_runtime.h>
#include <hip/hip_bf16.h>

typedef __bf16  v16bf __attribute__((ext_vector_type(16)));
typedef __bf16  bf16x8 __attribute__((ext_vector_type(8)));
typedef float   v8f   __attribute__((ext_vector_type(8)));

// Tile configuration
#define BM 128          // workgroup M tile
#define BN 128          // workgroup N tile
#define BK 32           // K step (== WMMA K for bf16)
#define LDA 40          // LDS row stride in bf16 elements (32 + 8 pad -> 80B rows, 16B aligned chunks)
#define WAVES 8         // 256 threads
#define QBLK 128        // dequant block size

union V16U { v16bf v; bf16x8 h[2]; };

__global__ __launch_bounds__(256)
void linear_bf16_wmma_kernel(const float* __restrict__ x,
                             const float* __restrict__ w,
                             const float* __restrict__ scl,
                             float* __restrict__ out,
                             int M, int N, int K) {
    __shared__ __align__(16) __bf16 As[BM * LDA];
    __shared__ __align__(16) __bf16 Ws[BN * LDA];

    const int tid    = threadIdx.x;
    const int lane   = tid & 31;
    const int lr     = lane & 15;     // lane row within half-wave
    const int lh     = lane >> 4;     // half-wave select
    const int waveId = tid >> 5;
    const int wm     = waveId >> 1;   // 0..3 -> wave M offset
    const int wn     = waveId & 1;    // 0..1 -> wave N offset
    const int mBase  = wm * 32;       // wave tile: 32 (M) x 64 (N)
    const int nBase  = wn * 64;

    const int n0 = blockIdx.x * BN;
    const int m0 = blockIdx.y * BM;

    // Global staging assignment: each thread owns 16 consecutive k of one row
    const int ldRow = tid >> 1;            // 0..127
    const int ldCol = (tid & 1) * 16;      // 0 or 16

    const int kBlocks = K / QBLK;          // scale row length

    v8f acc[2][4];
#pragma unroll
    for (int i = 0; i < 2; ++i)
#pragma unroll
        for (int j = 0; j < 4; ++j)
            acc[i][j] = (v8f)(0.0f);

    for (int k0 = 0; k0 < K; k0 += BK) {
        // ---- Stage A (activations) ----
        const float4* xp = reinterpret_cast<const float4*>(
            x + (size_t)(m0 + ldRow) * K + k0 + ldCol);
        float4 a0 = xp[0], a1 = xp[1], a2 = xp[2], a3 = xp[3];

        // ---- Stage W (weights) with uniform block-scale dequant ----
        const float s = scl[(n0 >> 7) * kBlocks + (k0 >> 7)];
        const float4* wp = reinterpret_cast<const float4*>(
            w + (size_t)(n0 + ldRow) * K + k0 + ldCol);
        float4 b0 = wp[0], b1 = wp[1], b2 = wp[2], b3 = wp[3];

        // Prefetch next K slab while we convert/compute
        if (k0 + BK < K) {
            __builtin_prefetch((const char*)(xp) + BK * sizeof(float), 0, 0);
            __builtin_prefetch((const char*)(wp) + BK * sizeof(float), 0, 0);
        }

        bf16x8 aLo = { (__bf16)a0.x, (__bf16)a0.y, (__bf16)a0.z, (__bf16)a0.w,
                       (__bf16)a1.x, (__bf16)a1.y, (__bf16)a1.z, (__bf16)a1.w };
        bf16x8 aHi = { (__bf16)a2.x, (__bf16)a2.y, (__bf16)a2.z, (__bf16)a2.w,
                       (__bf16)a3.x, (__bf16)a3.y, (__bf16)a3.z, (__bf16)a3.w };
        bf16x8 wLo = { (__bf16)(b0.x*s), (__bf16)(b0.y*s), (__bf16)(b0.z*s), (__bf16)(b0.w*s),
                       (__bf16)(b1.x*s), (__bf16)(b1.y*s), (__bf16)(b1.z*s), (__bf16)(b1.w*s) };
        bf16x8 wHi = { (__bf16)(b2.x*s), (__bf16)(b2.y*s), (__bf16)(b2.z*s), (__bf16)(b2.w*s),
                       (__bf16)(b3.x*s), (__bf16)(b3.y*s), (__bf16)(b3.z*s), (__bf16)(b3.w*s) };

        __syncthreads();   // previous iteration's fragment reads complete

        {
            bf16x8* ad = reinterpret_cast<bf16x8*>(&As[ldRow * LDA + ldCol]);
            ad[0] = aLo; ad[1] = aHi;
            bf16x8* wd = reinterpret_cast<bf16x8*>(&Ws[ldRow * LDA + ldCol]);
            wd[0] = wLo; wd[1] = wHi;
        }

        __syncthreads();   // tiles visible to all waves

        // ---- Fragment loads per ISA 16-bit A/B VGPR striping ----
        // A (16x32, MxK): lanes 0-15 hold K=0..7 & 16..23; lanes 16-31 hold K=8..15 & 24..31
        V16U af[2];
#pragma unroll
        for (int mt = 0; mt < 2; ++mt) {
            const __bf16* base = &As[(mBase + mt * 16 + lr) * LDA];
            af[mt].h[0] = *reinterpret_cast<const bf16x8*>(base + lh * 8);
            af[mt].h[1] = *reinterpret_cast<const bf16x8*>(base + 16 + lh * 8);
        }
        // B (32x16, KxN): lanes 0-15 hold K=0..7 & 16..23 of col N=lane; lanes 16-31 hold K=8..15 & 24..31
        V16U bf[4];
#pragma unroll
        for (int nt = 0; nt < 4; ++nt) {
            const __bf16* base = &Ws[(nBase + nt * 16 + lr) * LDA];
            bf[nt].h[0] = *reinterpret_cast<const bf16x8*>(base + lh * 8);
            bf[nt].h[1] = *reinterpret_cast<const bf16x8*>(base + 16 + lh * 8);
        }

        // ---- 8x WMMA ----
#pragma unroll
        for (int mt = 0; mt < 2; ++mt)
#pragma unroll
            for (int nt = 0; nt < 4; ++nt)
                acc[mt][nt] = __builtin_amdgcn_wmma_f32_16x16x32_bf16(
                    false, af[mt].v, false, bf[nt].v,
                    (short)0, acc[mt][nt], false, false);
    }

    // ---- Epilogue: C/D layout — VGPR r, lanes 0-15: (M=r, N=lane); lanes 16-31: (M=r+8) ----
#pragma unroll
    for (int mt = 0; mt < 2; ++mt) {
#pragma unroll
        for (int nt = 0; nt < 4; ++nt) {
            const int mrow0 = m0 + mBase + mt * 16 + lh * 8;
            const int ncol  = n0 + nBase + nt * 16 + lr;
#pragma unroll
            for (int r = 0; r < 8; ++r)
                out[(size_t)(mrow0 + r) * N + ncol] = acc[mt][nt][r];
        }
    }
}

extern "C" void kernel_launch(void* const* d_in, const int* in_sizes, int n_in,
                              void* d_out, int out_size, void* d_ws, size_t ws_size,
                              hipStream_t stream) {
    const float* x   = (const float*)d_in[0];   // [B*S, K] f32
    const float* w   = (const float*)d_in[1];   // [N, K]   f32 (fp8-valued)
    const float* scl = (const float*)d_in[2];   // [N/128, K/128] f32
    float*       out = (float*)d_out;           // [B*S, N] f32

    // Reference shapes: D_IN = D_OUT = 4096, M = B*S = in_sizes[0]/K
    const int K = 4096;
    const int N = 4096;
    const int M = in_sizes[0] / K;              // 8192

    dim3 grid(N / BN, M / BM);                  // 32 x 64 workgroups
    linear_bf16_wmma_kernel<<<grid, 256, 0, stream>>>(x, w, scl, out, M, N, K);
}
